// BiLSTMClassifier_77841987273386
// MI455X (gfx1250) — matrix operand order repaired and checked by reading, not compiled, not run on hardware
//
#include <hip/hip_runtime.h>
#include <hip/hip_bf16.h>

// ---------------------------------------------------------------------------
// BiLSTM (2 layers, bidirectional, H=256) + mean-pool + FC, MI455X / gfx1250.
//
// Persistent WGs (2 dirs x 4 batch-groups), combined [x|h] x [W_ih;W_hh]
// step-GEMM in bf16 WMMA 16x16x32 with f32 accumulate, gate-interleaved
// weight columns so gate fusion is wave-local, cell state and pooling sums
// register-resident across all 1024 timesteps, double-buffered A h-region in
// LDS. Weight B-fragments stream from L2 every step via inline-asm
// global_load_b128 (defeats LICM-to-scratch) in a 4-deep software pipeline
// with explicit s_wait_loadcnt. Bias is fused into the GEMM through a
// constant-1.0 row of the A matrix (free pad row for L0, one extra K-tile
// for L1), so the gate stage is pure register math off the staged C tile.
// ---------------------------------------------------------------------------

#define Bsz 64
#define Tlen 1024
#define DIN 12
#define Hdim 256
#define Gdim 1024      // 4H
#define NTt 64         // N tiles of 16 gate-columns
#define NCls 17

typedef __attribute__((ext_vector_type(16))) __bf16        v16bf;
typedef __attribute__((ext_vector_type(8)))  float         v8f;
typedef __attribute__((ext_vector_type(4)))  unsigned int  u32x4;
struct U32x8 { u32x4 lo, hi; };

__device__ __forceinline__ unsigned short f2bf(float f) {
    unsigned int u = __float_as_uint(f);
    unsigned int r = u + 0x7fffu + ((u >> 16) & 1u);   // round-to-nearest-even
    return (unsigned short)(r >> 16);
}
__device__ __forceinline__ float sigf(float x) { return 1.f / (1.f + __expf(-x)); }

// A-fragment swizzle for 16x32 bf16 tile: element (m,k) -> per-lane-contiguous
// position so the GEMM reads 32 contiguous bytes at [lane*16]. Every aligned
// 8-element K-octet is 16 contiguous bytes (one b128).
__device__ __forceinline__ int a_swz(int m, int k) {
    int lane = (k & 8) ? (m + 16) : m;                 // K 8-15,24-31 -> lanes 16-31
    int pos  = (k & 16) ? (8 + (k & 7)) : (k & 7);     // K 16-31 -> VGPR 4-7
    return lane * 16 + pos;
}

// 32-byte weight-fragment load, volatile so it can never be hoisted out of the
// time loop (weights must stream from L2 each step, not round-trip scratch).
__device__ __forceinline__ void gload_frag(const unsigned short* p, u32x4& lo, u32x4& hi) {
    asm volatile("global_load_b128 %0, %2, off\n\t"
                 "global_load_b128 %1, %2, off offset:16"
                 : "=&v"(lo), "=&v"(hi)
                 : "v"(p)
                 : "memory");
}
// Wait until the fragment being consumed is resident. Register-tied ("+v") so
// the dependent WMMA cannot be scheduled above the wait. Loads retire in
// order, so waiting loadcnt<=2*(younger frags) retires the oldest fragment;
// interleaved compiler loads only make the wait stricter.
__device__ __forceinline__ void wait_frag(int n, u32x4& lo, u32x4& hi) {
    if (n <= 0)      asm volatile("s_wait_loadcnt 0x0" : "+v"(lo), "+v"(hi));
    else if (n <= 2) asm volatile("s_wait_loadcnt 0x2" : "+v"(lo), "+v"(hi));
    else if (n <= 4) asm volatile("s_wait_loadcnt 0x4" : "+v"(lo), "+v"(hi));
    else             asm volatile("s_wait_loadcnt 0x6" : "+v"(lo), "+v"(hi));
}

// ---------------------------------------------------------------------------
// Packing kernels: bf16, fragment-swizzled, gate-interleaved weights.
// Wc layout: [dir][nt][kt][512 bf16], tile element order = B-fragment order
// (lane = n + (k&8?16:0), pos = (k&7) + (k&16?8:0)). Column n' = 4*j + gate (i,f,g,o).
// K-row KIN carries the combined bias (A supplies a constant 1.0 there).
// ---------------------------------------------------------------------------
__global__ void pack_x_kernel(const float* __restrict__ x, unsigned short* __restrict__ xbf) {
    int i = blockIdx.x * blockDim.x + threadIdx.x;     // < B*T*32
    if (i >= Bsz * Tlen * 32) return;
    int k = i & 31, bt = i >> 5;
    float v = (k < DIN) ? x[bt * DIN + k] : (k == DIN ? 1.0f : 0.f);  // 1.0 -> bias row
    xbf[i] = f2bf(v);
}

template <int KIN, int KXT, int KHT>
__global__ void pack_w_kernel(const float* __restrict__ w_ih, const float* __restrict__ w_hh,
                              const float* __restrict__ b_ih, const float* __restrict__ b_hh,
                              unsigned short* __restrict__ wc) {
    constexpr int KT = KXT + KHT;
    int i = blockIdx.x * blockDim.x + threadIdx.x;
    if (i >= 2 * NTt * KT * 512) return;
    int idx  = i & 511;
    int kt   = (i >> 9) % KT;
    int nt   = ((i >> 9) / KT) % NTt;
    int dir  = i / (NTt * KT * 512);
    int lane = idx >> 4, pos = idx & 15;
    int k_in = (pos & 7) | ((lane & 16) >> 1) | ((pos & 8) << 1);   // invert B swizzle
    int kk   = kt * 32 + k_in;
    int ncol = nt * 16 + (lane & 15);
    int j = ncol >> 2, g = ncol & 3;                   // gate-interleaved columns
    int row = g * Hdim + j;                            // PyTorch gate order i,f,g,o
    constexpr int KX = KXT * 32;
    float v;
    if (kk < KIN)       v = w_ih[(dir * 4 * Hdim + row) * KIN + kk];
    else if (kk == KIN) v = b_ih[dir * 4 * Hdim + row] + b_hh[dir * 4 * Hdim + row];
    else if (kk < KX)   v = 0.f;
    else                v = w_hh[(dir * 4 * Hdim + row) * Hdim + (kk - KX)];
    wc[i] = f2bf(v);
}

// ---------------------------------------------------------------------------
// Persistent LSTM kernel. grid = 8 blocks (dir = blk>>2, batch-group = blk&3),
// 512 threads = 16 waves; wave w owns N-tiles w*4 .. w*4+3 (single M-tile =
// its batch-group's 16 rows). Per step: build Ax (b128 moves), barrier,
// pipelined WMMA-GEMM over K tiles, wave-local gate fusion, packed-b32 h
// writes into the other Ah buffer, barrier.
// ---------------------------------------------------------------------------
template <int KXT, int KHT, bool IS_L1>
__global__ __launch_bounds__(512)
void lstm_kernel(const unsigned short* __restrict__ wc,      // [2][NT][KT][512]
                 const unsigned short* __restrict__ xin,     // L0: xbf[B][T][32]; L1: h0[2][B][T][H]
                 unsigned short*       __restrict__ hout,    // L0: h0[2][B][T][H]
                 float*                __restrict__ poolsum) // L1: [B][2H] sums
{
    constexpr int KT = KXT + KHT;
    const int dir  = blockIdx.x >> 2;
    const int bg   = blockIdx.x & 3;                   // batch rows bg*16 .. +15
    const int tid  = threadIdx.x;
    const int wave = tid >> 5, lane = tid & 31;

    __shared__ __attribute__((aligned(32))) unsigned short Ax[KXT * 512];
    __shared__ __attribute__((aligned(32))) unsigned short Ah[2][KHT * 512];
    __shared__ __attribute__((aligned(16))) float Cst[16 * 256];

    const unsigned short* wcd = wc + (size_t)dir * (NTt * KT * 512);

    for (int i = tid; i < KHT * 512; i += 512) Ah[0][i] = 0;   // h_{-1} = 0
    if (IS_L1) {
        // constant bias K-tile (tile KXT-1): column k=512 is 1.0, rest 0
        int m = tid >> 5, kin = tid & 31;
        Ax[(KXT - 1) * 512 + a_swz(m, kin)] = (kin == 0) ? (unsigned short)0x3F80 : 0;
    }

    float cst[8], pool[8];
#pragma unroll
    for (int i = 0; i < 8; ++i) { cst[i] = 0.f; pool[i] = 0.f; }

    for (int t = 0; t < Tlen; ++t) {
        const int tc = dir ? (Tlen - 1 - t) : t;

        // ---- build x-part of A: b128 octet moves (swizzle keeps octets intact)
        if (!IS_L1) {
            if (tid < 64) {                                       // 16m x 4 octets
                int m = tid >> 2, oct = tid & 3;
                *(u32x4*)&Ax[a_swz(m, oct * 8)] =
                    *(const u32x4*)&xin[((bg * 16 + m) * Tlen + tc) * 32 + oct * 8];
            }
        } else {
#pragma unroll
            for (int r = 0; r < 2; ++r) {                         // 16m x 64 octets
                int o = tid + r * 512;                            // lanes j-consecutive
                int m = o >> 6, rest = o & 63;
                int d2 = rest >> 5, j8 = rest & 31;               // concat(fwd, bwd)
                int kcol = d2 * 256 + j8 * 8;
                *(u32x4*)&Ax[(kcol >> 5) * 512 + a_swz(m, kcol & 31)] =
                    *(const u32x4*)&xin[(((d2 * Bsz) + bg * 16 + m) * Tlen + tc) * Hdim + j8 * 8];
            }
        }
        __syncthreads();

        const unsigned short* AhR = Ah[t & 1];
        unsigned short*       AhW = Ah[(t + 1) & 1];
        float* cw = &Cst[wave * 256];

#pragma unroll
        for (int s = 0; s < 4; ++s) {
            const int nt = wave * 4 + s;
            const unsigned short* wt = wcd + (size_t)(nt * KT) * 512 + lane * 16;
            if (nt + 1 < NTt)                                     // L2 prefetch, no counter
                __builtin_prefetch(wcd + (size_t)((nt + 1) * KT) * 512 + lane * 16, 0, 1);

            // 4-fragment-deep pipelined weight stream from L2
            u32x4 bl[4], bh[4];
#pragma unroll
            for (int i = 0; i < 4 && i < KT; ++i)
                gload_frag(wt + (size_t)i * 512, bl[i], bh[i]);

            v8f acc = {};
#pragma unroll
            for (int kt = 0; kt < KT; ++kt) {
                const int rem = KT - 1 - kt;                      // younger frags in flight
                wait_frag(2 * (rem < 3 ? rem : 3), bl[kt & 3], bh[kt & 3]);
                v16bf b = __builtin_bit_cast(v16bf, U32x8{bl[kt & 3], bh[kt & 3]});
                v16bf a;
                if (kt < KXT) a = *reinterpret_cast<const v16bf*>(&Ax[kt * 512 + lane * 16]);
                else          a = *reinterpret_cast<const v16bf*>(&AhR[(kt - KXT) * 512 + lane * 16]);
                acc = __builtin_amdgcn_wmma_f32_16x16x32_bf16(
                        false, a, false, b, (short)0, acc, false, false);
                if (kt + 4 < KT)                                  // refill the slot just freed
                    gload_frag(wt + (size_t)(kt + 4) * 512, bl[kt & 3], bh[kt & 3]);
            }
            // loadcnt is 0 here: compiler-tracked loads below see a clean counter.

            // stage tile to wave-private LDS (cross-lane gate gather)
#pragma unroll
            for (int jv = 0; jv < 8; ++jv)
                cw[(jv + ((lane >> 4) << 3)) * 16 + (lane & 15)] = acc[jv];

            const int m_in = lane & 15;
            float hpair[2];
#pragma unroll
            for (int q = 0; q < 2; ++q) {
                int u  = ((lane >> 4) << 1) + q;                  // unit-in-tile 0..3
                float gi = cw[m_in * 16 + u * 4 + 0];             // bias already in GEMM
                float gf = cw[m_in * 16 + u * 4 + 1];
                float gg = cw[m_in * 16 + u * 4 + 2];
                float go = cw[m_in * 16 + u * 4 + 3];
                int ci = s * 2 + q;                               // constant: regs, not scratch
                float c = sigf(gf) * cst[ci] + sigf(gi) * tanhf(gg);
                cst[ci] = c;
                float h = sigf(go) * tanhf(c);
                hpair[q] = h;
                if (IS_L1) pool[ci] += h;
            }
            // packed b32 h writes: units j0, j0+1 are adjacent in swizzle & hout
            unsigned int hp = (unsigned int)f2bf(hpair[0]) | ((unsigned int)f2bf(hpair[1]) << 16);
            int j0 = nt * 4 + ((lane >> 4) << 1);
            *(unsigned int*)&AhW[(j0 >> 5) * 512 + a_swz(m_in, j0 & 31)] = hp;
            if (!IS_L1)
                *(unsigned int*)&hout[(((dir * Bsz) + bg * 16 + m_in) * Tlen + tc) * Hdim + j0] = hp;
        }
        __syncthreads();
    }

    if (IS_L1) {
        const int m_in = lane & 15;
#pragma unroll
        for (int s = 0; s < 4; ++s)
#pragma unroll
            for (int q = 0; q < 2; ++q) {
                int u = ((lane >> 4) << 1) + q;
                int j = (wave * 4 + s) * 4 + u;
                poolsum[(bg * 16 + m_in) * (2 * Hdim) + dir * Hdim + j] = pool[s * 2 + q];
            }
    }
}

// ---------------------------------------------------------------------------
__global__ void fc_kernel(const float* __restrict__ poolsum, const float* __restrict__ fc_w,
                          const float* __restrict__ fc_b, float* __restrict__ out) {
    int i = blockIdx.x * blockDim.x + threadIdx.x;
    if (i >= Bsz * NCls) return;
    int b = i / NCls, c = i % NCls;
    float s = 0.f;
    for (int k = 0; k < 2 * Hdim; ++k)
        s += poolsum[b * 2 * Hdim + k] * fc_w[c * 2 * Hdim + k];
    out[i] = s / (float)Tlen + fc_b[c];
}

// ---------------------------------------------------------------------------
extern "C" void kernel_launch(void* const* d_in, const int* in_sizes, int n_in,
                              void* d_out, int out_size, void* d_ws, size_t ws_size,
                              hipStream_t stream) {
    const float* x     = (const float*)d_in[0];
    const float* w_ih0 = (const float*)d_in[1];
    const float* w_hh0 = (const float*)d_in[2];
    const float* b_ih0 = (const float*)d_in[3];
    const float* b_hh0 = (const float*)d_in[4];
    const float* w_ih1 = (const float*)d_in[5];
    const float* w_hh1 = (const float*)d_in[6];
    const float* b_ih1 = (const float*)d_in[7];
    const float* b_hh1 = (const float*)d_in[8];
    const float* fc_w  = (const float*)d_in[9];
    const float* fc_b  = (const float*)d_in[10];
    float* out = (float*)d_out;
    char*  ws  = (char*)d_ws;

    // workspace layout (bytes, 256-aligned); total ~75.9 MB
    const size_t off_wc0  = 0;                          // 2*64*9*512*2  = 1,179,648
    const size_t off_wc1  = 1179648;                    // 2*64*25*512*2 = 3,276,800
    const size_t off_xbf  = 4456448;                    // 64*1024*32*2  = 4,194,304
    const size_t off_h0   = 8650752;                    // 2*64*1024*256*2 = 67,108,864
    const size_t off_pool = 75759616;                   // 64*512*4 = 131,072

    unsigned short* wc0  = (unsigned short*)(ws + off_wc0);
    unsigned short* wc1  = (unsigned short*)(ws + off_wc1);
    unsigned short* xbf  = (unsigned short*)(ws + off_xbf);
    unsigned short* h0   = (unsigned short*)(ws + off_h0);
    float*          pool = (float*)(ws + off_pool);

    // pack / convert (deterministic, every element written incl. pads & bias rows)
    pack_x_kernel<<<(Bsz * Tlen * 32 + 255) / 256, 256, 0, stream>>>(x, xbf);
    pack_w_kernel<DIN, 1, 8><<<(2 * NTt * 9 * 512 + 255) / 256, 256, 0, stream>>>(
        w_ih0, w_hh0, b_ih0, b_hh0, wc0);
    pack_w_kernel<512, 17, 8><<<(2 * NTt * 25 * 512 + 255) / 256, 256, 0, stream>>>(
        w_ih1, w_hh1, b_ih1, b_hh1, wc1);

    // layer 0 (K = 32 pad/bias + 256), then layer 1 (K = 512 + 32 bias + 256)
    lstm_kernel<1, 8, false><<<8, 512, 0, stream>>>(wc0, xbf, h0, pool);
    lstm_kernel<17, 8, true><<<8, 512, 0, stream>>>(wc1, h0, h0, pool);

    // mean-pool (sums already in `pool`) + FC
    fc_kernel<<<(Bsz * NCls + 255) / 256, 256, 0, stream>>>(pool, fc_w, fc_b, out);

    (void)in_sizes; (void)n_in; (void)out_size; (void)ws_size;
}
